// LSTM_13864154431850
// MI455X (gfx1250) — matrix operand order repaired and checked by reading, not compiled, not run on hardware
//
#include <hip/hip_runtime.h>
#include <cstdint>
#include <cstddef>

// Problem constants (match reference).
#define T_SEQ  512
#define BATCH  128
#define IN_DIM 512
#define HID    512
#define NGATE  (4 * HID)   // 2048

typedef __attribute__((ext_vector_type(16))) __bf16        v16bf;
typedef __attribute__((ext_vector_type(8)))  float         v8f;
typedef __attribute__((ext_vector_type(8)))  unsigned int  v8u;
typedef __attribute__((ext_vector_type(4)))  int           v4i;

// float -> bf16 round-to-nearest-even
static __device__ __forceinline__ unsigned short f2bf(float f) {
    unsigned int u = __float_as_uint(f);
    unsigned int r = u + 0x7FFFu + ((u >> 16) & 1u);
    return (unsigned short)(r >> 16);
}

// Build a 16xbf16 WMMA fragment from two 16-byte contiguous chunks.
// (CDNA5 A/B bf16 lane layouts are 8 consecutive K-elements per b128.)
static __device__ __forceinline__ v16bf load_frag(const unsigned short* p0,
                                                  const unsigned short* p1) {
    uint4 a = *(const uint4*)p0;
    uint4 b = *(const uint4*)p1;
    v8u u;
    u[0] = a.x; u[1] = a.y; u[2] = a.z; u[3] = a.w;
    u[4] = b.x; u[5] = b.y; u[6] = b.z; u[7] = b.w;
    return __builtin_bit_cast(v16bf, u);
}

// Async global -> LDS copy of 16 bytes (gfx1250 GLOBAL_LOAD_ASYNC_TO_LDS_B128,
// tracked by ASYNCcnt). Builtin signature: (global v4i*, local v4i*, imm, imm).
static __device__ __forceinline__ void async_ld_b128(void* lds, const void* g) {
#if __has_builtin(__builtin_amdgcn_global_load_async_to_lds_b128)
    __builtin_amdgcn_global_load_async_to_lds_b128(
        (__attribute__((address_space(1))) v4i*)(void*)g,
        (__attribute__((address_space(3))) v4i*)lds, 0, 0);
#else
    unsigned lo = (unsigned)(uintptr_t)(__attribute__((address_space(3))) void*)lds;
    asm volatile("global_load_async_to_lds_b128 %0, %1, off"
                 :: "v"(lo), "v"(g) : "memory");
#endif
}

static __device__ __forceinline__ void wait_asynccnt0() {
#if __has_builtin(__builtin_amdgcn_s_wait_asynccnt)
    __builtin_amdgcn_s_wait_asynccnt(0);
#else
    asm volatile("s_wait_asynccnt 0" ::: "memory");
#endif
}

__global__ void cvt_f32_bf16(const float* __restrict__ in,
                             unsigned short* __restrict__ out, int n) {
    int i = blockIdx.x * blockDim.x + threadIdx.x;
    int stride = gridDim.x * blockDim.x;
    for (; i < n; i += stride) out[i] = f2bf(in[i]);
}

__global__ void bias_sum(const float* __restrict__ bx,
                         const float* __restrict__ bh,
                         float* __restrict__ bb, int n) {
    int i = blockIdx.x * blockDim.x + threadIdx.x;
    if (i < n) bb[i] = bx[i] + bh[i];
}

__global__ void zero_state(unsigned short* __restrict__ h0,
                           unsigned short* __restrict__ h1,
                           float* __restrict__ c) {
    int i = blockIdx.x * blockDim.x + threadIdx.x;   // n = BATCH*HID = 65536
    h0[i] = 0;
    h1[i] = 0;
    c[i]  = 0.0f;
}

// One LSTM timestep, fused: gates = x_t @ Wx^T + h_{t-1} @ Wh^T + (bx+bh),
// then elementwise cell update.
// Grid: 4 batch-supertiles (M=32) x 32 hid-chunks (N=16) = 128 WGs.
// Block: 128 threads = 4 wave32s; wave g computes gate g's two 16x16 tiles.
// A operands (x_t rows + h rows) staged through LDS with async-to-LDS copies
// in two K-phases of 256; B operands (Wx/Wh rows) streamed from L2.
__global__ __launch_bounds__(128)
void lstm_step(const unsigned short* __restrict__ xbf,   // [T,B,IN]  bf16
               const unsigned short* __restrict__ Wxbf,  // [4H,IN]   bf16
               const unsigned short* __restrict__ Whbf,  // [4H,H]    bf16
               const float*          __restrict__ bb,    // [4H]      f32
               const unsigned short* __restrict__ hprev, // [B,H]     bf16
               unsigned short*       __restrict__ hnext, // [B,H]     bf16
               float*                __restrict__ c,     // [B,H]     f32
               float*                __restrict__ out,   // [T,B,H]   f32
               int t)
{
    __shared__ __align__(16) unsigned short Ax[32][256];  // x_t rows, K-phase slice
    __shared__ __align__(16) unsigned short Ah[32][256];  // h rows,  K-phase slice
    __shared__ float gl[4][32][16];                       // gate exchange

    const int tid  = threadIdx.x;
    const int lane = tid & 31;
    const int gate = tid >> 5;                // 0..3 : i, f, g, o
    const int m0   = (blockIdx.x & 3) * 32;   // batch supertile origin
    const int n0   = (blockIdx.x >> 2) * 16;  // hidden chunk origin
    const int mn   = lane & 15;               // A row / B-C column within tile
    const int hi   = lane >> 4;               // lane half select

    const int r0 = gate * HID + n0;           // gate-row base into Wx/Wh

    // Accumulators seeded with bias (bias depends only on the column).
    const float bias = bb[r0 + mn];
    v8f acc0, acc1;
#pragma unroll
    for (int v = 0; v < 8; ++v) { acc0[v] = bias; acc1[v] = bias; }

    const unsigned short* xrow = xbf + (size_t)t * BATCH * IN_DIM;
    const unsigned short* Brow_x = Wxbf + (size_t)(r0 + mn) * IN_DIM;
    const unsigned short* Brow_h = Whbf + (size_t)(r0 + mn) * HID;

    // Warm L2 for next timestep's x rows (gfx1250 global_prefetch path).
    if (t + 1 < T_SEQ)
        __builtin_prefetch(xrow + (size_t)BATCH * IN_DIM + (size_t)(m0 + mn) * IN_DIM, 0, 1);

#pragma unroll
    for (int p = 0; p < 2; ++p) {
        const int k0 = p * 256;

        // ---- Stage A slices into LDS (async, shared by all 4 waves) ----
        // 32 rows x 256 cols per source; 8 bf16 (16B) per transfer.
        for (int idx = tid; idx < 1024; idx += 128) {
            const int r   = idx >> 5;          // 0..31
            const int col = (idx & 31) << 3;   // 0,8,...,248
            async_ld_b128(&Ax[r][col], xrow  + (size_t)(m0 + r) * IN_DIM + k0 + col);
            async_ld_b128(&Ah[r][col], hprev + (size_t)(m0 + r) * HID    + k0 + col);
        }
        wait_asynccnt0();
        __syncthreads();

        // ---- WMMA over this K-phase: x-projection + recurrent halves ----
#pragma unroll 2
        for (int kb = 0; kb < 256; kb += 32) {
            // x half
            v16bf ax0 = load_frag(&Ax[mn][kb + 8 * hi],      &Ax[mn][kb + 16 + 8 * hi]);
            v16bf ax1 = load_frag(&Ax[16 + mn][kb + 8 * hi], &Ax[16 + mn][kb + 16 + 8 * hi]);
            v16bf bx_ = load_frag(Brow_x + k0 + kb + 8 * hi, Brow_x + k0 + kb + 16 + 8 * hi);
            acc0 = __builtin_amdgcn_wmma_f32_16x16x32_bf16(false, ax0, false, bx_,
                                                           (short)0, acc0, false, false);
            acc1 = __builtin_amdgcn_wmma_f32_16x16x32_bf16(false, ax1, false, bx_,
                                                           (short)0, acc1, false, false);
            // h half
            v16bf ah0 = load_frag(&Ah[mn][kb + 8 * hi],      &Ah[mn][kb + 16 + 8 * hi]);
            v16bf ah1 = load_frag(&Ah[16 + mn][kb + 8 * hi], &Ah[16 + mn][kb + 16 + 8 * hi]);
            v16bf bh_ = load_frag(Brow_h + k0 + kb + 8 * hi, Brow_h + k0 + kb + 16 + 8 * hi);
            acc0 = __builtin_amdgcn_wmma_f32_16x16x32_bf16(false, ah0, false, bh_,
                                                           (short)0, acc0, false, false);
            acc1 = __builtin_amdgcn_wmma_f32_16x16x32_bf16(false, ah1, false, bh_,
                                                           (short)0, acc1, false, false);
        }
        __syncthreads();   // all waves done reading this phase's LDS before restage
    }

    // Exchange the 4 gate tiles through LDS (C/D layout: VGPR v -> row v+8*hi).
#pragma unroll
    for (int v = 0; v < 8; ++v) {
        gl[gate][v + 8 * hi][mn]      = acc0[v];
        gl[gate][16 + v + 8 * hi][mn] = acc1[v];
    }
    __syncthreads();

    // Elementwise cell update: 32x16 = 512 elements, 128 threads -> 4 each.
#pragma unroll
    for (int e = tid; e < 512; e += 128) {
        const int row = e >> 4, col = e & 15;
        float gi = gl[0][row][col];
        float gf = gl[1][row][col];
        float gg = gl[2][row][col];
        float go = gl[3][row][col];
        gi = 1.0f / (1.0f + __expf(-gi));
        gf = 1.0f / (1.0f + __expf(-gf));
        gg = tanhf(gg);
        go = 1.0f / (1.0f + __expf(-go));

        const size_t ci = (size_t)(m0 + row) * HID + (n0 + col);
        const float cn = c[ci] * gf + gi * gg;
        c[ci] = cn;
        const float hn = go * tanhf(cn);
        out[((size_t)t * BATCH + (m0 + row)) * (size_t)HID + (n0 + col)] = hn;
        hnext[ci] = f2bf(hn);
    }
}

extern "C" void kernel_launch(void* const* d_in, const int* in_sizes, int n_in,
                              void* d_out, int out_size, void* d_ws, size_t ws_size,
                              hipStream_t stream) {
    const float* x  = (const float*)d_in[0];   // [T,B,IN]
    const float* Wx = (const float*)d_in[1];   // [4H,IN]
    const float* bx = (const float*)d_in[2];   // [4H]
    const float* Wh = (const float*)d_in[3];   // [4H,H]
    const float* bh = (const float*)d_in[4];   // [4H]
    float* out = (float*)d_out;                // [T,B,H] f32

    // Workspace layout (all offsets 256B-aligned).
    char* ws = (char*)d_ws;
    size_t off = 0;
    unsigned short* xbf  = (unsigned short*)(ws + off); off += (size_t)T_SEQ * BATCH * IN_DIM * 2; // 64 MB
    unsigned short* Wxbf = (unsigned short*)(ws + off); off += (size_t)NGATE * IN_DIM * 2;         // 2 MB
    unsigned short* Whbf = (unsigned short*)(ws + off); off += (size_t)NGATE * HID * 2;            // 2 MB
    float*          bb   = (float*)(ws + off);          off += (size_t)NGATE * 4;
    unsigned short* h0   = (unsigned short*)(ws + off); off += (size_t)BATCH * HID * 2;
    unsigned short* h1   = (unsigned short*)(ws + off); off += (size_t)BATCH * HID * 2;
    float*          c    = (float*)(ws + off);          off += (size_t)BATCH * HID * 4;

    // One-time (per call) conversions: f32 -> bf16, bias fold, state zeroing.
    cvt_f32_bf16<<<4096, 256, 0, stream>>>(x,  xbf,  T_SEQ * BATCH * IN_DIM);
    cvt_f32_bf16<<<1024, 256, 0, stream>>>(Wx, Wxbf, NGATE * IN_DIM);
    cvt_f32_bf16<<<1024, 256, 0, stream>>>(Wh, Whbf, NGATE * HID);
    bias_sum<<<NGATE / 256, 256, 0, stream>>>(bx, bh, bb, NGATE);
    zero_state<<<(BATCH * HID) / 256, 256, 0, stream>>>(h0, h1, c);

    // Sequential recurrence: one fused kernel per timestep, h double-buffered.
    for (int t = 0; t < T_SEQ; ++t) {
        const unsigned short* hr = (t & 1) ? h1 : h0;
        unsigned short*       hw = (t & 1) ? h0 : h1;
        lstm_step<<<(BATCH / 32) * (HID / 16), 128, 0, stream>>>(
            xbf, Wxbf, Whbf, bb, hr, hw, c, out, t);
    }
}
